// YOWOv3_19756849562312
// MI455X (gfx1250) — compile-verified, matrix operand order checked by reading, not run on hardware
//
#include <hip/hip_runtime.h>
#include <hip/hip_bf16.h>
#include <stdint.h>

#define BATCH 32
#define NCLS 80
#define MANCH 6300
#define PADN 8192
#define TOPK_N 1000
#define TILES ((MANCH + 15) / 16)   // 394
#define IMGSZ 320.0f
#define CONF_TH 0.05f
#define NMS_TH 0.6f
#define EPSF 1e-10f

typedef __attribute__((ext_vector_type(16))) _Float16 v16h;
typedef __attribute__((ext_vector_type(8)))  float    v8f;

// ---- CDNA5 async global->LDS path (guarded: falls back to sync loads) -----
#if defined(__has_builtin)
#if __has_builtin(__builtin_amdgcn_global_load_async_to_lds_b64)
#define HAVE_ASYNC_LDS64 1
#endif
#if __has_builtin(__builtin_amdgcn_global_load_async_to_lds_b128)
#define HAVE_ASYNC_LDS128 1
#endif
#endif

// Parameter types per hipcc diagnostics:
//   arg0: int-vector pointer of the transfer width (global side)
//   arg1: int-vector pointer of the transfer width in LDS (AS3) ("__shared__")
typedef int avi2 __attribute__((vector_size(8)));
typedef int avi4 __attribute__((vector_size(16)));
typedef __attribute__((address_space(3))) avi2* lavi2;
typedef __attribute__((address_space(3))) avi4* lavi4;

__device__ __forceinline__ void wait_async0() {
#if defined(__has_builtin) && __has_builtin(__builtin_amdgcn_s_wait_asynccnt)
    __builtin_amdgcn_s_wait_asynccnt(0);
#else
    asm volatile("s_wait_asynccnt 0" ::: "memory");
#endif
}

__device__ __forceinline__ float sigm(float x) { return 1.0f / (1.0f + __expf(-x)); }

// Fill 8 consecutive A-operand elements (one 8-class run) with exp(x - mx), or zeros.
template <int OFF>
__device__ __forceinline__ void exp8(v16h& a, const float* __restrict__ p, float mx, bool valid) {
    if (valid) {
        float4 v0 = *(const float4*)(p);
        float4 v1 = *(const float4*)(p + 4);
        a[OFF + 0] = (_Float16)__expf(v0.x - mx);
        a[OFF + 1] = (_Float16)__expf(v0.y - mx);
        a[OFF + 2] = (_Float16)__expf(v0.z - mx);
        a[OFF + 3] = (_Float16)__expf(v0.w - mx);
        a[OFF + 4] = (_Float16)__expf(v1.x - mx);
        a[OFF + 5] = (_Float16)__expf(v1.y - mx);
        a[OFF + 6] = (_Float16)__expf(v1.z - mx);
        a[OFF + 7] = (_Float16)__expf(v1.w - mx);
    } else {
#pragma unroll
        for (int e = 0; e < 8; ++e) a[OFF + e] = (_Float16)0.0f;
    }
}

// ---------------------------------------------------------------------------
// Kernel 0: zero the padded sort-key arrays (pad entries sort to the bottom).
// ---------------------------------------------------------------------------
__global__ void yowo_init_keys(unsigned long long* __restrict__ keys) {
    int i = blockIdx.x * blockDim.x + threadIdx.x;
    if (i < BATCH * PADN) keys[i] = 0ull;
}

// ---------------------------------------------------------------------------
// Kernel 1: fused max/argmax + WMMA sum-exp + score. One wave = 16 anchors.
// Lane (r = lane&15, h = lane>>4). Each half-row scans 40 classes for max,
// partner-combined via shfl_xor(16). Sum of exp(x-max) over 80 classes is a
// 16x32 @ 32x16(ones) WMMA, 3 K-tiles (classes 80..95 zero-padded).
// Emits keys[b][m] = (score_bits << 32) | ~m  and labels[b][m].
// ---------------------------------------------------------------------------
__global__ void yowo_score_kernel(const float* __restrict__ conf,
                                  const float* __restrict__ cls,
                                  unsigned long long* __restrict__ keys,
                                  int* __restrict__ labels) {
    int wib  = threadIdx.x >> 5;
    int lane = threadIdx.x & 31;
    int r = lane & 15, h = lane >> 4;
    int gw   = blockIdx.x * (blockDim.x >> 5) + wib;
    int b    = gw / TILES;
    int tile = gw - b * TILES;
    bool wvalid = (b < BATCH);
    int bc = wvalid ? b : 0;
    int m  = tile * 16 + r;
    int mc = m < MANCH ? m : MANCH - 1;

    const float* row = cls + ((size_t)bc * MANCH + mc) * NCLS;
    __builtin_prefetch(conf + (size_t)bc * MANCH + mc, 0, 1);

    // ---- Phase A: per-half-row max/argmax over 40 classes, combine pairs ----
    float mx = -1e30f; int amx = 0;
    const float4* row4 = (const float4*)(row + h * 40);
#pragma unroll
    for (int q = 0; q < 10; ++q) {
        float4 v = row4[q];
        int cb = h * 40 + q * 4;
        if (v.x > mx) { mx = v.x; amx = cb; }
        if (v.y > mx) { mx = v.y; amx = cb + 1; }
        if (v.z > mx) { mx = v.z; amx = cb + 2; }
        if (v.w > mx) { mx = v.w; amx = cb + 3; }
    }
    float omx  = __shfl_xor(mx, 16);
    int   oamx = __shfl_xor(amx, 16);
    if (omx > mx || (omx == mx && oamx < amx)) { mx = omx; amx = oamx; }

    // ---- Phase B: sum exp(x - mx) over 80 classes via WMMA against ones ----
    v8f acc = {};
    v16h ones;
#pragma unroll
    for (int e = 0; e < 16; ++e) ones[e] = (_Float16)1.0f;

#pragma unroll
    for (int kt = 0; kt < 3; ++kt) {
        int base = kt * 32;
        int run0 = base + h * 8;        // A elems 0..7  <-> K = base+h*8 .. +7
        int run1 = base + 16 + h * 8;   // A elems 8..15 <-> K = base+16+h*8 ..
        v16h a;
        exp8<0>(a, row + run0, mx, run0 < NCLS);
        exp8<8>(a, row + run1, mx, run1 < NCLS);
        acc = __builtin_amdgcn_wmma_f32_16x16x32_f16(
            false, a, false, ones, (short)0, acc, false, false);
    }

    // ---- Extract row sums via LDS (C/D layout: lane0->rows0..7, lane16->8..15)
    __shared__ float s_sum[8][16];
    if (lane == 0) {
#pragma unroll
        for (int i = 0; i < 8; ++i) s_sum[wib][i] = acc[i];
    }
    if (lane == 16) {
#pragma unroll
        for (int i = 0; i < 8; ++i) s_sum[wib][8 + i] = acc[i];
    }
    __syncthreads();
    float sume = s_sum[wib][r];

    // ---- Phase C: score = sigmoid(conf)/sumexp; pack stable descending key --
    if (h == 0 && wvalid && m < MANCH) {
        float cf = conf[(size_t)b * MANCH + m];
        float score = sigm(cf) / sume;               // in (0,1): bits monotonic
        unsigned int kb = __float_as_uint(score);
        keys[(size_t)b * PADN + m] =
            ((unsigned long long)kb << 32) | (unsigned int)(~m);
        labels[(size_t)b * MANCH + m] = amx;
    }
}

// ---------------------------------------------------------------------------
// Kernel 2: per-batch descending bitonic sort of 8192 u64 keys in 64KB LDS.
// Initial fill uses CDNA5 async global->LDS DMA when available.
// ---------------------------------------------------------------------------
__global__ void yowo_sort_kernel(unsigned long long* __restrict__ keys) {
    __shared__ unsigned long long s[PADN];
    int b = blockIdx.x;
    unsigned long long* kb = keys + (size_t)b * PADN;
#if defined(HAVE_ASYNC_LDS64)
    for (int i = threadIdx.x; i < PADN; i += blockDim.x)
        __builtin_amdgcn_global_load_async_to_lds_b64(
            (avi2*)(kb + i), (lavi2)(&s[i]), 0, 0);
    wait_async0();
#else
    for (int i = threadIdx.x; i < PADN; i += blockDim.x) s[i] = kb[i];
#endif
    __syncthreads();
    for (int k = 2; k <= PADN; k <<= 1) {
        for (int j = k >> 1; j > 0; j >>= 1) {
            for (int t = threadIdx.x; t < PADN / 2; t += blockDim.x) {
                int i   = 2 * t - (t & (j - 1));
                int ixj = i | j;
                unsigned long long a = s[i], c = s[ixj];
                bool up = (i & k) == 0;                 // descending overall
                if (up ? (a < c) : (a > c)) { s[i] = c; s[ixj] = a; }
            }
            __syncthreads();
        }
    }
    for (int i = threadIdx.x; i < 1024; i += blockDim.x) kb[i] = s[i];
}

// ---------------------------------------------------------------------------
// Kernel 3: gather top-1000, decode boxes from reg/anchors, clip, write out.
// out layout: scores[32][1000] | labels[32][1000] | boxes[32][1000][4] | keep
// ---------------------------------------------------------------------------
__global__ void yowo_gather_kernel(const unsigned long long* __restrict__ keys,
                                   const int* __restrict__ labels,
                                   const float* __restrict__ reg,
                                   const float* __restrict__ anchors,
                                   const float* __restrict__ stridemap,
                                   float* __restrict__ out) {
    int b = blockIdx.y;
    int i = blockIdx.x * blockDim.x + threadIdx.x;
    if (i >= TOPK_N) return;
    unsigned long long key = keys[(size_t)b * PADN + i];
    int   m     = (int)(~(unsigned int)key);
    float score = __uint_as_float((unsigned int)(key >> 32));

    float4 rg = ((const float4*)reg)[(size_t)b * MANCH + m];
    float4 an = ((const float4*)anchors)[m];
    float  st = stridemap[m];
    float cx = an.x + sigm(rg.x) * st;
    float cy = an.y + sigm(rg.y) * st;
    float w  = an.z * __expf(rg.z);
    float h  = an.w * __expf(rg.w);
    float x1 = fminf(fmaxf((cx - 0.5f * w) / IMGSZ, 0.0f), 1.0f);
    float y1 = fminf(fmaxf((cy - 0.5f * h) / IMGSZ, 0.0f), 1.0f);
    float x2 = fminf(fmaxf((cx + 0.5f * w) / IMGSZ, 0.0f), 1.0f);
    float y2 = fminf(fmaxf((cy + 0.5f * h) / IMGSZ, 0.0f), 1.0f);

    out[(size_t)b * TOPK_N + i]         = score;
    out[32000 + (size_t)b * TOPK_N + i] = (float)labels[(size_t)b * MANCH + m];
    float4* ob = (float4*)(out + 64000);
    ob[(size_t)b * TOPK_N + i] = make_float4(x1, y1, x2, y2);
}

// ---------------------------------------------------------------------------
// Kernel 4: build NMS suppression bitmask. Block = 64 rows; columns staged in
// LDS 64 at a time (async global->LDS b128 when available). mask[b][i][w]
// bit jj set iff j=w*64+jj > i, same label, IoU > 0.6 (reference formula).
// ---------------------------------------------------------------------------
__global__ void yowo_nms_mask_kernel(const float* __restrict__ out,
                                     unsigned long long* __restrict__ mask) {
    int b = blockIdx.y;
    int i = blockIdx.x * 64 + threadIdx.x;
    const float4* boxes = (const float4*)(out + 64000) + (size_t)b * TOPK_N;
    const float*  labs  = out + 32000 + (size_t)b * TOPK_N;

    float4 bi = (i < TOPK_N) ? boxes[i] : make_float4(0, 0, 0, 0);
    float  li = (i < TOPK_N) ? labs[i] : -2.0f;
    float  ai = (bi.z - bi.x) * (bi.w - bi.y);

    __shared__ float4 sb[64];
    __shared__ float  sl[64];
    for (int jw = 0; jw < 16; ++jw) {
        int j0 = jw * 64;
        int jl = j0 + threadIdx.x;
#if defined(HAVE_ASYNC_LDS128)
        if (jl < TOPK_N) {
            __builtin_amdgcn_global_load_async_to_lds_b128(
                (avi4*)(boxes + jl), (lavi4)(&sb[threadIdx.x]), 0, 0);
        } else {
            sb[threadIdx.x] = make_float4(0, 0, 0, 0);
        }
        sl[threadIdx.x] = (jl < TOPK_N) ? labs[jl] : -3.0f;
        wait_async0();
#else
        sb[threadIdx.x] = (jl < TOPK_N) ? boxes[jl] : make_float4(0, 0, 0, 0);
        sl[threadIdx.x] = (jl < TOPK_N) ? labs[jl] : -3.0f;
#endif
        __syncthreads();
        unsigned long long bits = 0ull;
        if (i < TOPK_N) {
            for (int jj = 0; jj < 64; ++jj) {
                int j = j0 + jj;
                if (j < TOPK_N && j > i && sl[jj] == li) {
                    float4 bj = sb[jj];
                    float aj  = (bj.z - bj.x) * (bj.w - bj.y);
                    float xx1 = fmaxf(bi.x, bj.x), yy1 = fmaxf(bi.y, bj.y);
                    float xx2 = fminf(bi.z, bj.z), yy2 = fminf(bi.w, bj.w);
                    float inter = fmaxf(EPSF, xx2 - xx1) * fmaxf(EPSF, yy2 - yy1);
                    float uni   = fmaxf(ai + aj - inter, EPSF);
                    if (inter / uni > NMS_TH) bits |= (1ull << jj);
                }
            }
            mask[((size_t)b * TOPK_N + i) * 16 + jw] = bits;
        }
        __syncthreads();
    }
}

// ---------------------------------------------------------------------------
// Kernel 5: sequential keep scan per batch (1000 iterations, 16-word removed
// bitmask in LDS). keep[i] = !removed[i] && score>=0.05; kept boxes OR their
// suppression row into removed.
// ---------------------------------------------------------------------------
__global__ void yowo_nms_scan_kernel(const float* __restrict__ out_scores,
                                     const unsigned long long* __restrict__ mask,
                                     float* __restrict__ out_keep) {
    int b = blockIdx.x;
    __shared__ unsigned long long removed[16];
    if (threadIdx.x < 16) removed[threadIdx.x] = 0ull;
    __syncthreads();
    for (int i = 0; i < TOPK_N; ++i) {
        bool rem = (removed[i >> 6] >> (i & 63)) & 1ull;
        float sc = out_scores[(size_t)b * TOPK_N + i];
        bool keep = (!rem) && (sc >= CONF_TH);
        if (threadIdx.x == 0) out_keep[(size_t)b * TOPK_N + i] = keep ? 1.0f : 0.0f;
        __syncthreads();
        if (keep && threadIdx.x < 16)
            removed[threadIdx.x] |= mask[((size_t)b * TOPK_N + i) * 16 + threadIdx.x];
        __syncthreads();
    }
}

// ---------------------------------------------------------------------------
extern "C" void kernel_launch(void* const* d_in, const int* in_sizes, int n_in,
                              void* d_out, int out_size, void* d_ws, size_t ws_size,
                              hipStream_t stream) {
    const float* conf      = (const float*)d_in[0];
    const float* cls       = (const float*)d_in[1];
    const float* reg       = (const float*)d_in[2];
    const float* anchors   = (const float*)d_in[3];
    const float* stridemap = (const float*)d_in[4];
    float* out = (float*)d_out;

    char* ws = (char*)d_ws;
    unsigned long long* keys = (unsigned long long*)ws;                         // 2 MB
    int* labels = (int*)(ws + (size_t)BATCH * PADN * 8);                        // 806 KB
    unsigned long long* mask =
        (unsigned long long*)(ws + (size_t)BATCH * PADN * 8 + (size_t)BATCH * MANCH * 4); // 4 MB

    // 0) zero padded key arrays
    yowo_init_keys<<<(BATCH * PADN + 255) / 256, 256, 0, stream>>>(keys);

    // 1) fused score (WMMA sum-exp), one wave per 16 anchors, 8 waves/block
    int total_waves = BATCH * TILES;
    yowo_score_kernel<<<(total_waves + 7) / 8, 256, 0, stream>>>(conf, cls, keys, labels);

    // 2) per-batch bitonic sort in LDS (async global->LDS fill)
    yowo_sort_kernel<<<BATCH, 1024, 0, stream>>>(keys);

    // 3) gather + decode + clip top-1000
    yowo_gather_kernel<<<dim3((TOPK_N + 255) / 256, BATCH), 256, 0, stream>>>(
        keys, labels, reg, anchors, stridemap, out);

    // 4) NMS suppression bitmask
    yowo_nms_mask_kernel<<<dim3((TOPK_N + 63) / 64, BATCH), 64, 0, stream>>>(out, mask);

    // 5) sequential keep scan
    yowo_nms_scan_kernel<<<BATCH, 64, 0, stream>>>(out, mask, out + 192000);
}
